// GaussianBasis_41283225649437
// MI455X (gfx1250) — compile-verified, matrix operand order checked by reading, not run on hardware
//
#include <hip/hip_runtime.h>

typedef __attribute__((ext_vector_type(16))) _Float16 v16h;
typedef __attribute__((ext_vector_type(8)))  _Float16 v8h;
typedef __attribute__((ext_vector_type(8)))  float    v8f;
typedef __attribute__((ext_vector_type(4)))  float    v4f;

#define HH     256
#define WW     256
#define NPIX   65536        // H*W
#define NG     1024         // gaussians (K dim)
#define MM     50
#define RROWS  150          // M*3 output rows
#define RPAD   160          // padded to 10 tiles of 16
#define KCH    32           // WMMA K per step (f16)
#define NCHUNK (NG / KCH)   // 32
#define RT     (RPAD / 16)  // 10 row tiles

// ---------------------------------------------------------------------------
// Kernel 1: per-gaussian params -> packed {cx, cy, c1, c2, c3, op, 0, 0}
// ---------------------------------------------------------------------------
__global__ __launch_bounds__(256) void prep_params_kernel(
    const float* __restrict__ xyz_raw,
    const float* __restrict__ cholesky_raw,
    const float* __restrict__ opacity,
    float* __restrict__ params)
{
    int n = blockIdx.x * blockDim.x + threadIdx.x;
    if (n >= NG) return;
    float x  = tanhf(xyz_raw[2 * n + 0]);
    float y  = tanhf(xyz_raw[2 * n + 1]);
    float l1 = cholesky_raw[3 * n + 0] + 0.5f;
    float l2 = cholesky_raw[3 * n + 1];
    float l3 = cholesky_raw[3 * n + 2] + 0.5f;
    float a = l1 * l1;
    float b = l1 * l2;
    float c = l2 * l2 + l3 * l3;
    float inv = 1.0f / (a * c - b * b);
    float cx = 0.5f * ((x + 1.0f) * (float)WW - 1.0f);
    float cy = 0.5f * ((y + 1.0f) * (float)HH - 1.0f);
    float* p = params + n * 8;
    p[0] = cx;       p[1] = cy;
    p[2] = c * inv;  p[3] = -b * inv;  p[4] = a * inv;
    p[5] = opacity[n];
    p[6] = 0.0f;     p[7] = 0.0f;
}

// ---------------------------------------------------------------------------
// Kernel 2: features_dc[cluster_id] (f32 [M,N,3]) -> f16 A matrix [RPAD][NG]
//           row r = m*3 + c ; rows 150..159 zero padded
// ---------------------------------------------------------------------------
__global__ __launch_bounds__(256) void prep_feats_kernel(
    const float* __restrict__ feats,
    const int*  __restrict__ cluster_id,
    _Float16*   __restrict__ A16)
{
    int idx = blockIdx.x * blockDim.x + threadIdx.x;
    if (idx >= RPAD * NG) return;
    int r = idx >> 10;          // / NG
    int n = idx & (NG - 1);
    float v = 0.0f;
    if (r < RROWS) {
        int cid = cluster_id[0];
        int m = r / 3;
        int c = r - 3 * m;
        v = feats[(((size_t)cid * MM + m) * NG + n) * 3 + c];
    }
    A16[idx] = (_Float16)v;
}

// ---------------------------------------------------------------------------
// Kernel 3: fused weight-gen + GEMM via v_wmma_f32_16x16x32_f16.
// 4 waves / block, each wave owns one 16-pixel tile and all 10 row tiles.
// ---------------------------------------------------------------------------
__global__ __launch_bounds__(128) void gsum_wmma_kernel(
    const float*    __restrict__ params,
    const _Float16* __restrict__ A16,
    float*          __restrict__ out)
{
    __shared__ float sp[NG * 8];          // 32 KB gaussian params

    // cooperative LDS fill (b128 chunks)
    for (int i = threadIdx.x; i < (NG * 8) / 4; i += blockDim.x)
        ((v4f*)sp)[i] = ((const v4f*)params)[i];
    __syncthreads();

    const int lane = threadIdx.x & 31;
    const int wid  = threadIdx.x >> 5;
    const int col  = lane & 15;           // pixel within tile / row within A tile
    const int kgrp = lane >> 4;           // k half-group selector
    const int tile = blockIdx.x * 4 + wid;
    const int p0   = tile * 16;
    const float pxf = (float)((p0 + col) & (WW - 1));
    const float pyf = (float)((p0 + col) >> 8);

    v8f acc[RT] = {};

    for (int kc = 0; kc < NCHUNK; ++kc) {
        // ---- build B tile (32 gaussians x 16 pixels) in registers ----
        // CDNA5 16-bit B layout: lane<16 -> k in {0..7, 16..23}; lane>=16 -> {8..15, 24..31}
        v16h bmat;
        #pragma unroll
        for (int h = 0; h < 16; ++h) {
            int k = kc * KCH + ((h < 8) ? (kgrp * 8 + h)
                                        : (16 + kgrp * 8 + (h - 8)));
            v4f q0 = *(const v4f*)(sp + k * 8);       // cx, cy, c1, c2
            v4f q1 = *(const v4f*)(sp + k * 8 + 4);   // c3, op, -, -
            float dx = q0[0] - pxf;
            float dy = q0[1] - pyf;
            float sigma = 0.5f * (q0[2] * dx * dx + q1[0] * dy * dy)
                        + q0[3] * dx * dy;
            float alpha = fminf(0.999f, q1[1] * __expf(-sigma));
            float w = (sigma >= 0.0f && alpha >= (1.0f / 255.0f)) ? alpha : 0.0f;
            bmat[h] = (_Float16)w;
        }

        // ---- 10 WMMAs reuse this B tile (A streams from L2) ----
        const _Float16* abase = A16 + (size_t)kc * KCH + kgrp * 8;
        #pragma unroll
        for (int t = 0; t < RT; ++t) {
            const _Float16* ap = abase + (size_t)(t * 16 + col) * NG;
            v8h alo = *(const v8h*)(ap);        // k  0.. 7 (+kgrp*8)
            v8h ahi = *(const v8h*)(ap + 16);   // k 16..23 (+kgrp*8)
            v16h amat;
            #pragma unroll
            for (int i = 0; i < 8; ++i) { amat[i] = alo[i]; amat[i + 8] = ahi[i]; }
            acc[t] = __builtin_amdgcn_wmma_f32_16x16x32_f16(
                false, amat, false, bmat, (short)0, acc[t], false, false);
        }
    }

    // ---- store: C/D layout: VGPR v -> M = v (lanes 0-15) / v+8 (lanes 16-31)
    #pragma unroll
    for (int t = 0; t < RT; ++t) {
        #pragma unroll
        for (int v = 0; v < 8; ++v) {
            int row = t * 16 + v + (kgrp ? 8 : 0);
            if (row < RROWS)
                out[(size_t)row * NPIX + p0 + col] = acc[t][v];
        }
    }
}

// ---------------------------------------------------------------------------
extern "C" void kernel_launch(void* const* d_in, const int* in_sizes, int n_in,
                              void* d_out, int out_size, void* d_ws, size_t ws_size,
                              hipStream_t stream) {
    const float* xyz   = (const float*)d_in[0];
    const float* chol  = (const float*)d_in[1];
    const float* opac  = (const float*)d_in[2];
    const float* feats = (const float*)d_in[3];
    const int*   cid   = (const int*)d_in[4];
    float* out = (float*)d_out;

    float*    params = (float*)d_ws;                                   // 32 KB
    _Float16* A16    = (_Float16*)((char*)d_ws + NG * 8 * sizeof(float)); // 320 KB

    prep_params_kernel<<<(NG + 255) / 256, 256, 0, stream>>>(xyz, chol, opac, params);
    prep_feats_kernel<<<(RPAD * NG + 255) / 256, 256, 0, stream>>>(feats, cid, A16);

    // 4096 pixel tiles / 4 waves per block
    gsum_wmma_kernel<<<NPIX / 16 / 4, 128, 0, stream>>>(params, A16, out);
}
